// AGCRN_37529424233022
// MI455X (gfx1250) — matrix operand order repaired and checked by reading, not compiled, not run on hardware
//
#include <hip/hip_runtime.h>

// ---------------------------------------------------------------------------
// Problem constants (from reference)
// ---------------------------------------------------------------------------
#define N_NODES 10000
#define N_EDGES 100000
#define T_STEPS 8
#define F_IN    64
#define F_EDGE  16
#define F_OUT   128
#define NROWS   (T_STEPS * N_NODES)   // 80000 rows of x / Hn / agg
#define NCOLS   256                   // [R-gate 128 | update 128]

// Workspace layout (bytes, 256-aligned)
#define OFF_DEG   0u                        // 10000 f32
#define OFF_NORM  40960u                    // 100000 f32
#define OFF_WBN   441344u                   // 2*16*32*16 bf16 = 32768 B
#define OFF_WBE   474112u                   // 1*16*32*16 bf16 = 16384 B
#define OFF_H     490496u                   // 80000*256 bf16 = 40,960,000 B
#define OFF_EH    41450496u                 // 100000*256 bf16 = 51,200,000 B
#define OFF_AGG   92650496u                 // 80000*256 f32  = 81,920,000 B

typedef __attribute__((ext_vector_type(16))) __bf16          v16bf;
typedef __attribute__((ext_vector_type(8)))  float           v8f;
typedef __attribute__((ext_vector_type(16))) unsigned short  v16u;

__device__ __forceinline__ unsigned short f2bf(float f) {
    unsigned u = __builtin_bit_cast(unsigned, f);
    unsigned r = u + 0x7FFFu + ((u >> 16) & 1u);   // round-to-nearest-even
    return (unsigned short)(r >> 16);
}
__device__ __forceinline__ float bf2f(unsigned short b) {
    return __builtin_bit_cast(float, (unsigned)b << 16);
}

// ---------------------------------------------------------------------------
// 0) zero degree + accumulator
// ---------------------------------------------------------------------------
__global__ void k_zero(float* __restrict__ deg, float* __restrict__ agg) {
    const long long na = (long long)NROWS * NCOLS;
    for (long long i = blockIdx.x * 256LL + threadIdx.x; i < na;
         i += (long long)gridDim.x * 256LL) {
        agg[i] = 0.0f;
        if (i < N_NODES) deg[i] = 0.0f;
    }
}

// ---------------------------------------------------------------------------
// 1) pack weight columns into WMMA B-fragment layout (bf16)
//    B 32x16 tile: lane l holds column (l&15), elems j<8 -> K = (l>>4)*8 + j, j>=8 -> K = 16 + (l>>4)*8 + (j-8)
//    Column mapping: N<128 -> Wn_gate[:, 128+N] (R gate), else Wn_upd[:, N-128]
// ---------------------------------------------------------------------------
__global__ void k_pack(const float* __restrict__ Wng, const float* __restrict__ Wnu,
                       const float* __restrict__ Weg, const float* __restrict__ Weu,
                       unsigned short* __restrict__ WBn, unsigned short* __restrict__ WBe) {
    int idx = blockIdx.x * 256 + threadIdx.x;          // 96 blocks -> 24576 threads
    if (idx < 16384) {                                 // node weights: [kt=2][nt=16][lane=32][j=16]
        int j    = idx & 15;
        int lane = (idx >> 4) & 31;
        int nt   = (idx >> 9) & 15;
        int kt   = idx >> 13;
        int K = kt * 32 + (j < 8 ? (lane >> 4) * 8 + j : 16 + (lane >> 4) * 8 + (j - 8));        // 0..63
        int N = nt * 16 + (lane & 15);                 // 0..255
        float v = (N < 128) ? Wng[K * 256 + 128 + N] : Wnu[K * 128 + (N - 128)];
        WBn[idx] = f2bf(v);
    } else if (idx < 24576) {                          // edge weights: [nt=16][lane=32][j=16], K padded
        int i2   = idx - 16384;
        int j    = i2 & 15;
        int lane = (i2 >> 4) & 31;
        int nt   = (i2 >> 9) & 15;
        int K = (j < 8 ? (lane >> 4) * 8 + j : 16 + (lane >> 4) * 8 + (j - 8));                  // 0..31, valid rows 0..15
        int N = nt * 16 + (lane & 15);
        float v = 0.0f;
        if (K < F_EDGE) v = (N < 128) ? Weg[K * 256 + 128 + N] : Weu[K * 128 + (N - 128)];
        WBe[i2] = f2bf(v);
    }
}

// ---------------------------------------------------------------------------
// 2) degree count + symmetric norm
// ---------------------------------------------------------------------------
__global__ void k_deg(const int* __restrict__ ei, float* __restrict__ deg) {
    int e = blockIdx.x * 256 + threadIdx.x;
    if (e < N_EDGES) atomicAdd(&deg[ei[N_EDGES + e]], 1.0f);
}

__global__ void k_norm(const int* __restrict__ ei, const float* __restrict__ deg,
                       float* __restrict__ norm) {
    int e = blockIdx.x * 256 + threadIdx.x;
    if (e < N_EDGES) {
        float ds = fmaxf(deg[ei[e]], 1.0f);
        float dd = fmaxf(deg[ei[N_EDGES + e]], 1.0f);
        norm[e] = __frsqrt_rn(ds * dd);
    }
}

// ---------------------------------------------------------------------------
// A-fragment builders (16x32 bf16 tile from f32 row-major source)
// lane l: row = l&15; elems 0..7 -> K = kbase..kbase+7, elems 8..15 -> K = kbase+16..kbase+23
// ---------------------------------------------------------------------------
__device__ __forceinline__ v16bf a_frag_f32(const float* __restrict__ rowptr, int kbase) {
    v16u u;
    const float4 f0 = *(const float4*)(rowptr + kbase);
    const float4 f1 = *(const float4*)(rowptr + kbase + 4);
    const float4 g0 = *(const float4*)(rowptr + kbase + 16);
    const float4 g1 = *(const float4*)(rowptr + kbase + 20);
    u[0]  = f2bf(f0.x); u[1]  = f2bf(f0.y); u[2]  = f2bf(f0.z); u[3]  = f2bf(f0.w);
    u[4]  = f2bf(f1.x); u[5]  = f2bf(f1.y); u[6]  = f2bf(f1.z); u[7]  = f2bf(f1.w);
    u[8]  = f2bf(g0.x); u[9]  = f2bf(g0.y); u[10] = f2bf(g0.z); u[11] = f2bf(g0.w);
    u[12] = f2bf(g1.x); u[13] = f2bf(g1.y); u[14] = f2bf(g1.z); u[15] = f2bf(g1.w);
    return __builtin_bit_cast(v16bf, u);
}

__device__ __forceinline__ v16bf a_frag_edge(const float* __restrict__ rowptr, int kbase) {
    v16u u;
    const float4 f0 = *(const float4*)(rowptr + kbase);
    const float4 f1 = *(const float4*)(rowptr + kbase + 4);
    u[0] = f2bf(f0.x); u[1] = f2bf(f0.y); u[2] = f2bf(f0.z); u[3] = f2bf(f0.w);
    u[4] = f2bf(f1.x); u[5] = f2bf(f1.y); u[6] = f2bf(f1.z); u[7] = f2bf(f1.w);
    u[8] = 0; u[9] = 0; u[10] = 0; u[11] = 0; u[12] = 0; u[13] = 0; u[14] = 0; u[15] = 0;
    return __builtin_bit_cast(v16bf, u);
}

// ---------------------------------------------------------------------------
// 3) node GEMM: Hn[80000,256] = x[80000,64] @ Wcat[64,256], output bf16
//    block = 4 waves, one 16-row M tile per block, wave w covers N columns 64w..64w+63
// ---------------------------------------------------------------------------
__global__ void k_node_gemm(const float* __restrict__ x, const unsigned short* __restrict__ WB,
                            unsigned short* __restrict__ Hn) {
    const int mtile = blockIdx.x;                 // 0..4999
    const int wave  = threadIdx.x >> 5;           // 0..3
    const int lane  = threadIdx.x & 31;
    const int m     = lane & 15;
    const int hi    = lane >> 4;
    const float* xr = x + (long long)(mtile * 16 + m) * F_IN;

    v16bf a0 = a_frag_f32(xr, 0 * 32 + hi * 8);
    v16bf a1 = a_frag_f32(xr, 1 * 32 + hi * 8);

#pragma unroll
    for (int nt4 = 0; nt4 < 4; ++nt4) {
        const int nt = wave * 4 + nt4;
        v8f c = {};
        {
            v16u bu = *(const v16u*)(WB + ((0 * 16 + nt) * 32 + lane) * 16);
            v16bf b = __builtin_bit_cast(v16bf, bu);
            c = __builtin_amdgcn_wmma_f32_16x16x32_bf16(false, a0, false, b, (short)0, c, false, false);
        }
        {
            v16u bu = *(const v16u*)(WB + ((1 * 16 + nt) * 32 + lane) * 16);
            v16bf b = __builtin_bit_cast(v16bf, bu);
            c = __builtin_amdgcn_wmma_f32_16x16x32_bf16(false, a1, false, b, (short)0, c, false, false);
        }
        // C layout: lane l holds column nt*16+(l&15), rows (l>>4)*8 + r
        const int ncol = nt * 16 + m;
#pragma unroll
        for (int r = 0; r < 8; ++r) {
            long long row = (long long)(mtile * 16 + hi * 8 + r);
            Hn[row * NCOLS + ncol] = f2bf(c[r]);
        }
    }
}

// ---------------------------------------------------------------------------
// 4) edge GEMM: EH[100000,256] = edge_attr[100000,16] @ Wecat[16,256], output bf16
// ---------------------------------------------------------------------------
__global__ void k_edge_gemm(const float* __restrict__ ea, const unsigned short* __restrict__ WB,
                            unsigned short* __restrict__ EH) {
    const int mtile = blockIdx.x;                 // 0..6249
    const int wave  = threadIdx.x >> 5;
    const int lane  = threadIdx.x & 31;
    const int m     = lane & 15;
    const int hi    = lane >> 4;
    const float* er = ea + (long long)(mtile * 16 + m) * F_EDGE;

    v16bf a = a_frag_edge(er, hi * 8);

#pragma unroll
    for (int nt4 = 0; nt4 < 4; ++nt4) {
        const int nt = wave * 4 + nt4;
        v8f c = {};
        v16u bu = *(const v16u*)(WB + (nt * 32 + lane) * 16);
        v16bf b = __builtin_bit_cast(v16bf, bu);
        c = __builtin_amdgcn_wmma_f32_16x16x32_bf16(false, a, false, b, (short)0, c, false, false);
        const int ncol = nt * 16 + m;
#pragma unroll
        for (int r = 0; r < 8; ++r) {
            long long row = (long long)(mtile * 16 + hi * 8 + r);
            EH[row * NCOLS + ncol] = f2bf(c[r]);
        }
    }
}

// ---------------------------------------------------------------------------
// 5) message pass: one wave per edge; lane owns 8 consecutive channels.
//    agg[t*N+dst, c] += (Hn[t*N+src, c] + EH[e, c]) * norm[e]
// ---------------------------------------------------------------------------
__global__ void k_message(const int* __restrict__ ei, const float* __restrict__ norm,
                          const unsigned short* __restrict__ Hn,
                          const unsigned short* __restrict__ EH,
                          float* __restrict__ agg) {
    const int e = blockIdx.x * 8 + (threadIdx.x >> 5);   // 12500 blocks exact
    if (e >= N_EDGES) return;
    const int lane = threadIdx.x & 31;
    const int s = ei[e];
    const int d = ei[N_EDGES + e];
    const float nrm = norm[e];

    // EH row chunk (8 bf16 = 16B)
    float eh[8];
    {
        uint4 raw = *(const uint4*)(EH + (long long)e * NCOLS + lane * 8);
        const unsigned short* p = (const unsigned short*)&raw;
#pragma unroll
        for (int j = 0; j < 8; ++j) eh[j] = bf2f(p[j]);
    }

#pragma unroll
    for (int t = 0; t < T_STEPS; ++t) {
        uint4 raw = *(const uint4*)(Hn + ((long long)(t * N_NODES + s)) * NCOLS + lane * 8);
        const unsigned short* p = (const unsigned short*)&raw;
        float* dst = agg + ((long long)(t * N_NODES + d)) * NCOLS + lane * 8;
#pragma unroll
        for (int j = 0; j < 8; ++j) {
            atomicAdd(&dst[j], (bf2f(p[j]) + eh[j]) * nrm);
        }
    }
}

// ---------------------------------------------------------------------------
// 6) pointwise GRU combine: out = (1 - sigmoid(aggR + bR)) * tanh(aggU + bU)
// ---------------------------------------------------------------------------
__global__ void k_final(const float* __restrict__ agg, const float* __restrict__ b_gate,
                        const float* __restrict__ b_upd, float* __restrict__ out) {
    const long long idx = (long long)blockIdx.x * 256 + threadIdx.x;  // 40000 blocks exact
    const long long row = idx >> 7;          // t*N + n
    const int c = (int)(idx & 127);
    const float zr = agg[row * NCOLS + c]       + b_gate[128 + c];
    const float zu = agg[row * NCOLS + 128 + c] + b_upd[c];
    const float R = 1.0f / (1.0f + __expf(-zr));
    out[idx] = (1.0f - R) * tanhf(zu);
}

// ---------------------------------------------------------------------------
extern "C" void kernel_launch(void* const* d_in, const int* in_sizes, int n_in,
                              void* d_out, int out_size, void* d_ws, size_t ws_size,
                              hipStream_t stream) {
    (void)in_sizes; (void)n_in; (void)out_size; (void)ws_size;
    const float* X   = (const float*)d_in[0];   // [1,T,N,64]
    const int*   EI  = (const int*)d_in[1];     // [2,E]
    const float* EA  = (const float*)d_in[2];   // [E,16]
    const float* Wng = (const float*)d_in[3];   // [192,256]
    const float* Weg = (const float*)d_in[4];   // [16,256]
    const float* bg  = (const float*)d_in[5];   // [256]
    const float* Wnu = (const float*)d_in[6];   // [192,128]
    const float* Weu = (const float*)d_in[7];   // [16,128]
    const float* bu  = (const float*)d_in[8];   // [128]
    float* out = (float*)d_out;

    char* ws = (char*)d_ws;
    float*          deg  = (float*)(ws + OFF_DEG);
    float*          nrm  = (float*)(ws + OFF_NORM);
    unsigned short* WBn  = (unsigned short*)(ws + OFF_WBN);
    unsigned short* WBe  = (unsigned short*)(ws + OFF_WBE);
    unsigned short* Hn   = (unsigned short*)(ws + OFF_H);
    unsigned short* EH   = (unsigned short*)(ws + OFF_EH);
    float*          agg  = (float*)(ws + OFF_AGG);

    k_zero<<<4096, 256, 0, stream>>>(deg, agg);
    k_pack<<<96, 256, 0, stream>>>(Wng, Wnu, Weg, Weu, WBn, WBe);
    k_deg<<<(N_EDGES + 255) / 256, 256, 0, stream>>>(EI, deg);
    k_norm<<<(N_EDGES + 255) / 256, 256, 0, stream>>>(EI, deg, nrm);
    k_node_gemm<<<NROWS / 16, 128, 0, stream>>>(X, WBn, Hn);
    k_edge_gemm<<<N_EDGES / 16, 128, 0, stream>>>(EA, WBe, EH);
    k_message<<<N_EDGES / 8, 256, 0, stream>>>(EI, nrm, Hn, EH, agg);
    k_final<<<(T_STEPS * N_NODES * F_OUT) / 256, 256, 0, stream>>>(agg, bg, bu, out);
}
